// MixerMamba2_17059610100189
// MI455X (gfx1250) — matrix operand order repaired and checked by reading, not compiled, not run on hardware
//
#include <hip/hip_runtime.h>

// ---- problem constants ----
constexpr int kB    = 2;
constexpr int kL    = 2048;
constexpr int kDM   = 2048;
constexpr int kDS   = 128;
constexpr int kDC   = 4;
constexpr int kHD   = 64;
constexpr int kCH   = 256;
constexpr int kDI   = 4096;               // EXP*DM
constexpr int kNH   = 64;                 // DI/HD
constexpr int kDIP  = 8512;               // 2*DI + 2*DS + NH
constexpr int kCDIM = 4352;               // DI + 2*DS
constexpr int kNC   = kL / kCH;           // 8 chunks
constexpr int kBL   = kB * kL;            // 4096 rows
constexpr float kEPS = 1e-5f;

typedef __attribute__((ext_vector_type(2)))  float   v2f;
typedef __attribute__((ext_vector_type(8)))  float   v8f;
typedef __attribute__((ext_vector_type(16))) __bf16  v16bf;

static __device__ __forceinline__ v8f v8f_zero() {
  v8f z = {0.f,0.f,0.f,0.f,0.f,0.f,0.f,0.f};
  return z;
}
static __device__ __forceinline__ float sigmoidf_(float x) { return 1.0f / (1.0f + __expf(-x)); }
static __device__ __forceinline__ float siluf_(float x)    { return x * sigmoidf_(x); }

// ---- CDNA5 async global->LDS copies (ASYNCcnt-tracked) ----
static __device__ __forceinline__ void async_ld_b128(void* lds, const void* g) {
  uint32_t lo = (uint32_t)(uintptr_t)lds;          // flat LDS addr[31:0] == LDS offset
  asm volatile("global_load_async_to_lds_b128 %0, %1, off" :: "v"(lo), "v"(g) : "memory");
}
static __device__ __forceinline__ void async_ld_b32(void* lds, const void* g) {
  uint32_t lo = (uint32_t)(uintptr_t)lds;
  asm volatile("global_load_async_to_lds_b32 %0, %1, off" :: "v"(lo), "v"(g) : "memory");
}
static __device__ __forceinline__ void wait_async0() {
  asm volatile("s_wait_asynccnt 0x0" ::: "memory");
}

// split 16 contiguous-ish f32 (two octets) into bf16 hi/lo WMMA fragments
static __device__ __forceinline__ void split16(const float* __restrict__ s0,
                                               const float* __restrict__ s1,
                                               v16bf& hi, v16bf& lo) {
#pragma unroll
  for (int i = 0; i < 8; ++i) {
    float v0 = s0[i], v1 = s1[i];
    __bf16 h0 = (__bf16)v0, h1 = (__bf16)v1;
    hi[i]     = h0;
    hi[8 + i] = h1;
    lo[i]     = (__bf16)(v0 - (float)h0);
    lo[8 + i] = (__bf16)(v1 - (float)h1);
  }
}

// =====================================================================
// K1: zxbcdt = u @ w_in  (4096 x 8512 x 2048)
// bf16x3 split precision, async double-buffered LDS tiles.
// block 256 thr (8 waves), 128x128 tile, wave = 32x64 subtile.
// =====================================================================
__global__ void __launch_bounds__(256)
k1_gemm_in(const float* __restrict__ U, const float* __restrict__ W,
           float* __restrict__ ZX) {
  __shared__ __align__(16) float Af[2][128][32];   // A tile, natural [m][k]
  __shared__ __align__(16) float Bf[2][128][32];   // B tile, transposed [n][k]

  const int tid  = threadIdx.x;
  const int lane = tid & 31;
  const int wave = tid >> 5;
  const int wm   = (wave & 3) * 32;
  const int wn   = (wave >> 2) * 64;
  const int m0   = blockIdx.y * 128;
  const int n0   = blockIdx.x * 128;
  const int lm   = lane & 15, lh = lane >> 4;
  const int kbA  = lh * 8;    // A frag: lanes0-15 K 0..7/16..23, lanes16-31 K 8..15/24..31
  const int kbB  = lh * 8;    // B frag: lanes0-15 K 0..7/16..23, lanes16-31 K 8..15/24..31

  v8f acc[2][4];
#pragma unroll
  for (int i = 0; i < 2; ++i)
#pragma unroll
    for (int j = 0; j < 4; ++j) acc[i][j] = v8f_zero();

  auto issue_stage = [&](int buf, int kt) {
    // A: 128x32 f32 = 1024 x 16B chunks -> 32 b128 copies (4 per wave)
#pragma unroll
    for (int i = 0; i < 4; ++i) {
      int q  = ((wave * 4 + i) << 5) | lane;
      int r  = q >> 3, c4 = (q & 7) << 2;
      async_ld_b128(&Af[buf][r][c4], U + (size_t)(m0 + r) * kDM + kt + c4);
    }
    // B: 32x128 f32, transposing scatter -> 128 b32 copies (16 per wave)
#pragma unroll
    for (int i = 0; i < 16; ++i) {
      int d  = ((wave * 16 + i) << 5) | lane;
      int k  = d >> 7, n = d & 127;
      int gn = n0 + n; gn = (gn < kDIP) ? gn : (kDIP - 1);   // clamp OOB cols (never stored)
      async_ld_b32(&Bf[buf][n][k], W + (size_t)(kt + k) * kDIP + gn);
    }
  };

  issue_stage(0, 0);
  for (int kt = 0; kt < kDM; kt += 32) {
    const int buf = (kt >> 5) & 1;
    wait_async0();          // this wave's copies for `buf` complete
    __syncthreads();        // everyone's copies complete; prior buf reads done
    if (kt + 32 < kDM) issue_stage(buf ^ 1, kt + 32);

    v16bf ahi[2], alo[2];
#pragma unroll
    for (int mi = 0; mi < 2; ++mi) {
      const int mrow = wm + mi * 16 + lm;
      split16(&Af[buf][mrow][kbA], &Af[buf][mrow][16 + kbA], ahi[mi], alo[mi]);
    }
#pragma unroll
    for (int ni = 0; ni < 4; ++ni) {
      const int ncol = wn + ni * 16 + lm;
      v16bf bhi, blo;
      split16(&Bf[buf][ncol][kbB], &Bf[buf][ncol][16 + kbB], bhi, blo);
#pragma unroll
      for (int mi = 0; mi < 2; ++mi) {
        acc[mi][ni] = __builtin_amdgcn_wmma_f32_16x16x32_bf16(false, ahi[mi], false, bhi, (short)0, acc[mi][ni], false, false);
        acc[mi][ni] = __builtin_amdgcn_wmma_f32_16x16x32_bf16(false, ahi[mi], false, blo, (short)0, acc[mi][ni], false, false);
        acc[mi][ni] = __builtin_amdgcn_wmma_f32_16x16x32_bf16(false, alo[mi], false, bhi, (short)0, acc[mi][ni], false, false);
      }
    }
  }

#pragma unroll
  for (int mi = 0; mi < 2; ++mi)
#pragma unroll
    for (int ni = 0; ni < 4; ++ni) {
      const int rbase = m0 + wm + mi * 16 + lh * 8;
      const int col   = n0 + wn + ni * 16 + lm;
      if (col < kDIP) {
#pragma unroll
        for (int j = 0; j < 8; ++j)
          ZX[(size_t)(rbase + j) * kDIP + col] = acc[mi][ni][j];
      }
    }
}

// =====================================================================
// K2: dt = softplus(raw + dt_bias); dA = -exp(A_log)*dt   (layout b,h,l)
// =====================================================================
__global__ void __launch_bounds__(256)
k2_dt(const float* __restrict__ ZX, const float* __restrict__ DTB,
      const float* __restrict__ ALOG, float* __restrict__ DT, float* __restrict__ DA) {
  int idx = blockIdx.x * 256 + threadIdx.x;          // over BL*NH
  if (idx >= kBL * kNH) return;
  int h   = idx & (kNH - 1);
  int row = idx >> 6;                                // b*L + l
  float v  = ZX[(size_t)row * kDIP + kDI + kCDIM + h] + DTB[h];
  float dt = (v > 20.0f) ? v : log1pf(__expf(v));
  DT[(size_t)row * kNH + h] = dt;
  int b = row >> 11, l = row & (kL - 1);
  DA[((size_t)(b * kNH + h)) * kL + l] = -__expf(ALOG[h]) * dt;
}

// =====================================================================
// K3: causal depthwise conv (DC=4) + bias + SiLU; split x / B / C
// =====================================================================
__global__ void __launch_bounds__(256)
k3_conv(const float* __restrict__ ZX, const float* __restrict__ CW,
        const float* __restrict__ CB, float* __restrict__ XC,
        float* __restrict__ BC, float* __restrict__ CC) {
  const int row = blockIdx.x;            // b*L + l
  const int l   = row & (kL - 1);
  for (int c = threadIdx.x; c < kCDIM; c += 256) {
    float a = CB[c];
#pragma unroll
    for (int k = 0; k < kDC; ++k) {
      int lp = l + k - (kDC - 1);
      if (lp >= 0)
        a += ZX[(size_t)(row + k - (kDC - 1)) * kDIP + kDI + c] * CW[c * kDC + k];
    }
    a = siluf_(a);
    if (c < kDI)            XC[(size_t)row * kDI + c] = a;
    else if (c < kDI + kDS) BC[(size_t)row * kDS + (c - kDI)] = a;
    else                    CC[(size_t)row * kDS + (c - kDI - kDS)] = a;
  }
}

// =====================================================================
// K4: per-chunk inclusive cumsum of dA  (1 thread / (b,h,chunk))
// =====================================================================
__global__ void __launch_bounds__(256)
k4_scan(const float* __restrict__ DA, float* __restrict__ ACS) {
  int idx = blockIdx.x * 256 + threadIdx.x;          // B*NH*NC = 1024
  if (idx >= kB * kNH * kNC) return;
  int c = idx & (kNC - 1), h = (idx >> 3) & (kNH - 1), b = idx >> 9;
  size_t base = ((size_t)(b * kNH + h)) * kL + c * kCH;
  float r = 0.f;
  for (int t = 0; t < kCH; ++t) { r += DA[base + t]; ACS[base + t] = r; }
}

// =====================================================================
// K5: intra-chunk Yd. block = 64 thr (2 waves), one (b,c,h,rowblock32).
// S = (C@B^T) * exp(Acs[t]-Acs[s]) (tril), then Yd = S @ (x*dt).
// f32 WMMA 16x16x4 both stages; triangular pruning on K.
// =====================================================================
__global__ void __launch_bounds__(64)
k5_yd(const float* __restrict__ XC, const float* __restrict__ BC,
      const float* __restrict__ CC, const float* __restrict__ DT,
      const float* __restrict__ ACS, float* __restrict__ Y) {
  __shared__ float S[2][16][kCH];
  __shared__ float sA[kCH];
  __shared__ float sDt[kCH];

  const int tid = threadIdx.x, lane = tid & 31, w = tid >> 5;
  int bid = blockIdx.x;
  const int rb = bid & 7;            bid >>= 3;
  const int h  = bid & (kNH - 1);    bid >>= 6;
  const int c  = bid & (kNC - 1);    bid >>= 3;
  const int b  = bid;
  const int lbase = b * kL + c * kCH;
  const int abase = (b * kNH + h) * kL + c * kCH;

  for (int i = tid; i < kCH; i += 64) {
    sA[i]  = ACS[abase + i];
    sDt[i] = DT[(size_t)(lbase + i) * kNH + h];
  }
  __syncthreads();

  const int lm = lane & 15, lh = lane >> 4;
  const int t0 = rb * 32 + w * 16;
  const int ttile = t0 >> 4;

  // ---- phase A: score rows [t0, t0+16) x [0, (ttile+1)*16) ----
  const float* Crow = CC + (size_t)(lbase + t0 + lm) * kDS;
  for (int st = 0; st <= ttile; ++st) {
    v8f g = v8f_zero();
    const float* Brow = BC + (size_t)(lbase + st * 16 + lm) * kDS;
    for (int k0 = 0; k0 < kDS; k0 += 4) {
      int k = k0 + 2 * lh;
      v2f a; a.x = Crow[k]; a.y = Crow[k + 1];
      v2f bb; bb.x = Brow[k]; bb.y = Brow[k + 1];
      g = __builtin_amdgcn_wmma_f32_16x16x4_f32(false, a, false, bb, (short)0, g, false, false);
    }
    const int s = st * 16 + lm;
#pragma unroll
    for (int j = 0; j < 8; ++j) {
      int t = t0 + j + 8 * lh;
      float v = g[j];
      v = (s <= t) ? v * __expf(sA[t] - sA[s]) : 0.0f;
      S[w][j + 8 * lh][s] = v;
    }
  }
  // per-wave private S region: same-wave DS ordering is enough

  // ---- phase B: Yd(16x64) = S(16xkmax) @ xdt(kmax x 64) ----
  const int kmax = (ttile + 1) * 16;
#pragma unroll
  for (int ni = 0; ni < 4; ++ni) {
    v8f accv = v8f_zero();
    const int n = ni * 16 + lm;
    for (int k0 = 0; k0 < kmax; k0 += 4) {
      int k = k0 + 2 * lh;
      v2f a; a.x = S[w][lm][k]; a.y = S[w][lm][k + 1];
      v2f bb;
      bb.x = XC[(size_t)(lbase + k) * kDI + h * kHD + n] * sDt[k];
      bb.y = XC[(size_t)(lbase + k + 1) * kDI + h * kHD + n] * sDt[k + 1];
      accv = __builtin_amdgcn_wmma_f32_16x16x4_f32(false, a, false, bb, (short)0, accv, false, false);
    }
#pragma unroll
    for (int j = 0; j < 8; ++j) {
      int t = t0 + j + 8 * lh;
      Y[(size_t)(lbase + t) * kDI + h * kHD + n] = accv[j];
    }
  }
}

// =====================================================================
// K6: chunk states (HD x DS) = (x*dt*decay)^T @ B.  block=256, (b,c,h)
// =====================================================================
__global__ void __launch_bounds__(256)
k6_states(const float* __restrict__ XC, const float* __restrict__ BC,
          const float* __restrict__ DT, const float* __restrict__ ACS,
          float* __restrict__ ST) {
  __shared__ float sW[kCH];   // dt[t]*exp(Aend - Acs[t])
  const int tid = threadIdx.x, lane = tid & 31, w = tid >> 5;
  int bid = blockIdx.x;
  const int h = bid & (kNH - 1); bid >>= 6;
  const int c = bid & (kNC - 1); bid >>= 3;
  const int b = bid;
  const int lbase = b * kL + c * kCH;
  const int abase = (b * kNH + h) * kL + c * kCH;
  const float aend = ACS[abase + kCH - 1];
  sW[tid] = DT[(size_t)(lbase + tid) * kNH + h] * __expf(aend - ACS[abase + tid]);
  __syncthreads();

  const int lm = lane & 15, lh = lane >> 4;
  const size_t sbase = ((size_t)((b * kNC + c) * kNH + h)) * kHD * kDS;
#pragma unroll
  for (int i = 0; i < 4; ++i) {
    const int tile = w * 4 + i;          // 32 tiles: 4 (p) x 8 (n)
    const int p0 = (tile & 3) * 16, n0 = (tile >> 2) * 16;
    v8f accv = v8f_zero();
    for (int k0 = 0; k0 < kCH; k0 += 4) {
      int k = k0 + 2 * lh;
      v2f a;
      a.x = XC[(size_t)(lbase + k) * kDI + h * kHD + p0 + lm] * sW[k];
      a.y = XC[(size_t)(lbase + k + 1) * kDI + h * kHD + p0 + lm] * sW[k + 1];
      v2f bb;
      bb.x = BC[(size_t)(lbase + k) * kDS + n0 + lm];
      bb.y = BC[(size_t)(lbase + k + 1) * kDS + n0 + lm];
      accv = __builtin_amdgcn_wmma_f32_16x16x4_f32(false, a, false, bb, (short)0, accv, false, false);
    }
#pragma unroll
    for (int j = 0; j < 8; ++j)
      ST[sbase + (size_t)(p0 + j + 8 * lh) * kDS + n0 + lm] = accv[j];
  }
}

// =====================================================================
// K7: inter-chunk recurrence: init[0]=0; init[c+1]=exp(Asum[c])*init[c]+st[c]
// =====================================================================
__global__ void __launch_bounds__(256)
k7_rec(const float* __restrict__ ST, const float* __restrict__ ACS,
       float* __restrict__ IS) {
  __shared__ float sAs[kNC];
  const int bh = blockIdx.x;                 // b*NH + h
  const int h = bh & (kNH - 1), b = bh >> 6;
  if (threadIdx.x < kNC)
    sAs[threadIdx.x] = ACS[((size_t)(b * kNH + h)) * kL + threadIdx.x * kCH + kCH - 1];
  __syncthreads();
  for (int e = threadIdx.x; e < kHD * kDS; e += 256) {
    float R = 0.f;
    for (int c = 0; c < kNC; ++c) {
      size_t idx = ((size_t)((b * kNC + c) * kNH + h)) * kHD * kDS + e;
      IS[idx] = R;
      R = R * __expf(sAs[c]) + ST[idx];
    }
  }
}

// =====================================================================
// K8: Yo = exp(Acs[t]) * (C @ init_st^T), accumulate into Y. (b,c,h)
// =====================================================================
__global__ void __launch_bounds__(256)
k8_yo(const float* __restrict__ CC, const float* __restrict__ IS,
      const float* __restrict__ ACS, float* __restrict__ Y) {
  __shared__ float sA[kCH];
  const int tid = threadIdx.x, lane = tid & 31, w = tid >> 5;
  int bid = blockIdx.x;
  const int h = bid & (kNH - 1); bid >>= 6;
  const int c = bid & (kNC - 1); bid >>= 3;
  const int b = bid;
  const int lbase = b * kL + c * kCH;
  const int abase = (b * kNH + h) * kL + c * kCH;
  sA[tid] = ACS[abase + tid];
  __syncthreads();

  const int lm = lane & 15, lh = lane >> 4;
  const size_t isbase = ((size_t)((b * kNC + c) * kNH + h)) * kHD * kDS;
#pragma unroll
  for (int i = 0; i < 8; ++i) {
    const int tile = w * 8 + i;          // 64 tiles: 16 (t) x 4 (p)
    const int t0 = (tile >> 2) * 16, p0 = (tile & 3) * 16;
    v8f accv = v8f_zero();
    for (int k0 = 0; k0 < kDS; k0 += 4) {
      int k = k0 + 2 * lh;
      v2f a;
      a.x = CC[(size_t)(lbase + t0 + lm) * kDS + k];
      a.y = CC[(size_t)(lbase + t0 + lm) * kDS + k + 1];
      v2f bb;
      bb.x = IS[isbase + (size_t)(p0 + lm) * kDS + k];
      bb.y = IS[isbase + (size_t)(p0 + lm) * kDS + k + 1];
      accv = __builtin_amdgcn_wmma_f32_16x16x4_f32(false, a, false, bb, (short)0, accv, false, false);
    }
#pragma unroll
    for (int j = 0; j < 8; ++j) {
      int t = t0 + j + 8 * lh;
      size_t yi = (size_t)(lbase + t) * kDI + h * kHD + p0 + lm;
      Y[yi] = Y[yi] + accv[j] * __expf(sA[t]);
    }
  }
}

// =====================================================================
// K9: y += x*D; yg = y*silu(z); RMSNorm * norm_w.  block per (b,l) row.
// =====================================================================
__global__ void __launch_bounds__(256)
k9_final(const float* __restrict__ Y, const float* __restrict__ XC,
         const float* __restrict__ ZX, const float* __restrict__ DSK,
         const float* __restrict__ NW, float* __restrict__ OUT) {
  __shared__ float red[256];
  const int row = blockIdx.x, tid = threadIdx.x;
  float regs[16];
  float ss = 0.f;
#pragma unroll
  for (int i = 0; i < 16; ++i) {
    const int cidx = i * 256 + tid;
    float xv = XC[(size_t)row * kDI + cidx];
    float yv = Y[(size_t)row * kDI + cidx] + xv * DSK[cidx >> 6];
    float z  = ZX[(size_t)row * kDIP + cidx];
    float yg = yv * siluf_(z);
    regs[i] = yg;
    ss += yg * yg;
  }
  red[tid] = ss;
  __syncthreads();
  for (int s = 128; s > 0; s >>= 1) {
    if (tid < s) red[tid] += red[tid + s];
    __syncthreads();
  }
  const float scale = rsqrtf(red[0] / (float)kDI + kEPS);
#pragma unroll
  for (int i = 0; i < 16; ++i) {
    const int cidx = i * 256 + tid;
    OUT[(size_t)row * kDI + cidx] = regs[i] * scale * NW[cidx];
  }
}

// =====================================================================
extern "C" void kernel_launch(void* const* d_in, const int* in_sizes, int n_in,
                              void* d_out, int out_size, void* d_ws, size_t ws_size,
                              hipStream_t stream) {
  (void)in_sizes; (void)n_in; (void)out_size; (void)ws_size;
  const float* u       = (const float*)d_in[0];
  const float* w_in    = (const float*)d_in[1];
  const float* conv_w  = (const float*)d_in[2];
  const float* conv_b  = (const float*)d_in[3];
  const float* dt_bias = (const float*)d_in[4];
  const float* A_log   = (const float*)d_in[5];
  const float* D_skip  = (const float*)d_in[6];
  const float* norm_w  = (const float*)d_in[7];
  float* out = (float*)d_out;

  float* ws = (float*)d_ws;
  size_t off = 0;
  float* zx  = ws + off; off += (size_t)kBL * kDIP;                 // 139.5 MB
  float* xc  = ws + off; off += (size_t)kBL * kDI;                  //  67.1 MB
  float* bc  = ws + off; off += (size_t)kBL * kDS;
  float* cc  = ws + off; off += (size_t)kBL * kDS;
  float* dt  = ws + off; off += (size_t)kBL * kNH;
  float* da  = ws + off; off += (size_t)kBL * kNH;
  float* acs = ws + off; off += (size_t)kBL * kNH;
  float* st  = ws + off; off += (size_t)kB * kNC * kNH * kHD * kDS; //  33.6 MB
  float* ist = ws + off; off += (size_t)kB * kNC * kNH * kHD * kDS;
  float* yb  = ws + off; off += (size_t)kBL * kDI;

  dim3 g1((kDIP + 127) / 128, kBL / 128);
  k1_gemm_in<<<g1, 256, 0, stream>>>(u, w_in, zx);
  k2_dt<<<(kBL * kNH) / 256, 256, 0, stream>>>(zx, dt_bias, A_log, dt, da);
  k3_conv<<<kBL, 256, 0, stream>>>(zx, conv_w, conv_b, xc, bc, cc);
  k4_scan<<<(kB * kNH * kNC + 255) / 256, 256, 0, stream>>>(da, acs);
  k5_yd<<<kB * kNC * kNH * 8, 64, 0, stream>>>(xc, bc, cc, dt, acs, yb);
  k6_states<<<kB * kNC * kNH, 256, 0, stream>>>(xc, bc, dt, acs, st);
  k7_rec<<<kB * kNH, 256, 0, stream>>>(st, acs, ist);
  k8_yo<<<kB * kNC * kNH, 256, 0, stream>>>(cc, ist, acs, yb);
  k9_final<<<kBL, 256, 0, stream>>>(yb, xc, zx, D_skip, norm_w, out);
}
